// MAICDelayAgent_77189152244004
// MI455X (gfx1250) — matrix operand order repaired and checked by reading, not compiled, not run on hardware
//
#include <hip/hip_runtime.h>
#include <hip/hip_bf16.h>

typedef __attribute__((ext_vector_type(16))) _Float16 v16h;
typedef __attribute__((ext_vector_type(8)))  _Float16 v8h;
typedef __attribute__((ext_vector_type(8)))  float    v8f;

#define NROWS 32768      // BS*A
#define HD    64
#define IND   96
#define LD_   16
#define NNH_  64
#define ATT_  32
#define NACT_ 16
#define BS_   1024
#define AG    32
#define DLAT  512        // A*L
#define M2ROWS (BS_*AG*AG)

// ---------------- WMMA helpers (CDNA5 wave32, 16x16x32 f16 -> f32) ----------------

__device__ __forceinline__ v8f wmma16(v16h a, v16h b, v8f c) {
    return __builtin_amdgcn_wmma_f32_16x16x32_f16(false, a, false, b, (short)0, c, false, false);
}

// A fragment (16x32): row = row0 + (lane&15); lane-half selects K block.
// Element j<8 -> k = k0 + 8*(lane>=16) + j ; j>=8 -> +16. Two contiguous float4 pairs.
__device__ __forceinline__ v16h ldA_f32(const float* __restrict__ p, int ld, int row0, int k0, int lane) {
    const float* rp = p + (long)(row0 + (lane & 15)) * ld + k0 + ((lane >> 4) << 3);
    const float4 u0 = *(const float4*)(rp);
    const float4 u1 = *(const float4*)(rp + 4);
    const float4 u2 = *(const float4*)(rp + 16);
    const float4 u3 = *(const float4*)(rp + 20);
    v16h a;
    a[0] = (_Float16)u0.x; a[1] = (_Float16)u0.y; a[2] = (_Float16)u0.z; a[3] = (_Float16)u0.w;
    a[4] = (_Float16)u1.x; a[5] = (_Float16)u1.y; a[6] = (_Float16)u1.z; a[7] = (_Float16)u1.w;
    a[8]  = (_Float16)u2.x; a[9]  = (_Float16)u2.y; a[10] = (_Float16)u2.z; a[11] = (_Float16)u2.w;
    a[12] = (_Float16)u3.x; a[13] = (_Float16)u3.y; a[14] = (_Float16)u3.z; a[15] = (_Float16)u3.w;
    return a;
}

// A fragment for K=16 matrices (latent), zero-padded to K=32: only j<8 valid.
__device__ __forceinline__ v16h ldA_lat(const float* __restrict__ p, int row0, int lane) {
    const float* rp = p + (long)(row0 + (lane & 15)) * LD_ + ((lane >> 4) << 3);
    const float4 u0 = *(const float4*)(rp);
    const float4 u1 = *(const float4*)(rp + 4);
    v16h a;
    a[0] = (_Float16)u0.x; a[1] = (_Float16)u0.y; a[2] = (_Float16)u0.z; a[3] = (_Float16)u0.w;
    a[4] = (_Float16)u1.x; a[5] = (_Float16)u1.y; a[6] = (_Float16)u1.z; a[7] = (_Float16)u1.w;
#pragma unroll
    for (int j = 8; j < 16; ++j) a[j] = (_Float16)0.f;
    return a;
}

// A fragment from wave-local LDS f16 tile (16 x ld): two contiguous 16B reads.
__device__ __forceinline__ v16h ldA_lds(const _Float16* p, int ld, int k0, int lane) {
    const _Float16* rp = p + (lane & 15) * ld + k0 + ((lane >> 4) << 3);
    const v8h lo = *(const v8h*)(rp);
    const v8h hi = *(const v8h*)(rp + 16);
    v16h a;
#pragma unroll
    for (int j = 0; j < 8; ++j) { a[j] = lo[j]; a[j + 8] = hi[j]; }
    return a;
}

// B fragment (32x16) from pre-transposed f16 weights WT[k*nout + n]: lane&15 -> N,
// element j<8 -> k = k0 + 8*(lane>=16) + j ; j>=8 -> +16 (strided loads).
__device__ __forceinline__ v16h ldB_t(const _Float16* __restrict__ WT, int nout, int n0, int k0, int lane) {
    const _Float16* rp = WT + (long)(k0 + ((lane >> 4) << 3)) * nout + n0 + (lane & 15);
    v16h b;
#pragma unroll
    for (int j = 0; j < 8; ++j) { b[j] = rp[(long)j * nout]; b[j + 8] = rp[(long)(j + 16) * nout]; }
    return b;
}

// Bounded variant: rows k >= kin produce zeros (K=16 tail of msg_w1).
__device__ __forceinline__ v16h ldB_t_bk(const _Float16* __restrict__ WT, int nout, int n0, int k0, int kin, int lane) {
    v16h b;
    const int kb = k0 + ((lane >> 4) << 3);
    const _Float16* rp = WT + (long)kb * nout + n0 + (lane & 15);
#pragma unroll
    for (int j = 0; j < 8; ++j) {
        b[j] = (kb + j < kin) ? rp[(long)j * nout] : (_Float16)0.f;
        b[j + 8] = (kb + 16 + j < kin) ? rp[(long)(j + 16) * nout] : (_Float16)0.f;
    }
    return b;
}

// ---------------- K0: weight repack W(nout x kin, f32) -> WT f16 [k*nout + n] ----------------

__global__ void __launch_bounds__(256) k0_repack(const float* __restrict__ W, _Float16* __restrict__ WT,
                                                 int nout, int kin)
{
    const int e = blockIdx.x * 256 + threadIdx.x;
    if (e < nout * kin) {
        const int n = e / kin, k = e - n * kin;   // contiguous global reads along k
        WT[k * nout + n] = (_Float16)W[e];
    }
}

// f16 copy without transpose (for matrices already in B layout, e.g. wk_w[k][n])
__global__ void __launch_bounds__(256) k0_tof16(const float* __restrict__ W, _Float16* __restrict__ WH, int nelem)
{
    const int e = blockIdx.x * 256 + threadIdx.x;
    if (e < nelem) WH[e] = (_Float16)W[e];
}

// ---------------- K1: fc1 + GRU + q + z1 (fused, one 16-row tile per wave) ----------------

__global__ void __launch_bounds__(256) k1_fc1_gru(
    const float* __restrict__ inputs, const float* __restrict__ h_in,
    const _Float16* __restrict__ fc1T, const float* __restrict__ fc1_b,
    const _Float16* __restrict__ wihT, const _Float16* __restrict__ whhT,
    const float* __restrict__ bih, const float* __restrict__ bhh,
    const _Float16* __restrict__ fc2T, const float* __restrict__ fc2_b,
    const _Float16* __restrict__ emb1T, const float* __restrict__ emb_b1,
    float* __restrict__ out_q, float* __restrict__ out_h, float* __restrict__ z1_buf)
{
    const int lane = threadIdx.x & 31;
    const int wave = threadIdx.x >> 5;
    const int m0 = (blockIdx.x * 8 + wave) * 16;
    const int n = lane & 15;
    const int mh = (lane >> 4) << 3;

    // warm L2/WGP$ with the GRU weights (global_prefetch_b8)
    if (threadIdx.x < 192) {
        __builtin_prefetch(wihT + threadIdx.x * 64, 0, 1);
        __builtin_prefetch(whhT + threadIdx.x * 64, 0, 1);
    }

    __shared__ _Float16 sh[8][16 * HD];
    _Float16* xs = sh[wave];

    // x = relu(inputs @ fc1_w^T + fc1_b), staged to LDS as f16
    v16h ai0 = ldA_f32(inputs, IND, m0, 0, lane);
    v16h ai1 = ldA_f32(inputs, IND, m0, 32, lane);
    v16h ai2 = ldA_f32(inputs, IND, m0, 64, lane);
#pragma unroll
    for (int c = 0; c < 4; ++c) {
        v8f acc = {};
        acc = wmma16(ai0, ldB_t(fc1T, HD, c * 16, 0, lane), acc);
        acc = wmma16(ai1, ldB_t(fc1T, HD, c * 16, 32, lane), acc);
        acc = wmma16(ai2, ldB_t(fc1T, HD, c * 16, 64, lane), acc);
        const int col = c * 16 + n;
        const float bv = fc1_b[col];
#pragma unroll
        for (int r = 0; r < 8; ++r) {
            float v = acc[r] + bv;
            xs[(mh + r) * HD + col] = (_Float16)(v > 0.f ? v : 0.f);
        }
    }

    v16h xa0 = ldA_lds(xs, HD, 0, lane);
    v16h xa1 = ldA_lds(xs, HD, 32, lane);
    v16h ha0 = ldA_f32(h_in, HD, m0, 0, lane);
    v16h ha1 = ldA_f32(h_in, HD, m0, 32, lane);

    // GRU gates (r,z,n), 16 output columns at a time; h written to out + LDS (f16)
#pragma unroll
    for (int c = 0; c < 4; ++c) {
        v8f a_ir = {}, a_iz = {}, a_in = {}, a_hr = {}, a_hz = {}, a_hn = {};
        a_ir = wmma16(xa0, ldB_t(wihT, 192, c * 16, 0, lane), a_ir);
        a_ir = wmma16(xa1, ldB_t(wihT, 192, c * 16, 32, lane), a_ir);
        a_iz = wmma16(xa0, ldB_t(wihT, 192, 64 + c * 16, 0, lane), a_iz);
        a_iz = wmma16(xa1, ldB_t(wihT, 192, 64 + c * 16, 32, lane), a_iz);
        a_in = wmma16(xa0, ldB_t(wihT, 192, 128 + c * 16, 0, lane), a_in);
        a_in = wmma16(xa1, ldB_t(wihT, 192, 128 + c * 16, 32, lane), a_in);
        a_hr = wmma16(ha0, ldB_t(whhT, 192, c * 16, 0, lane), a_hr);
        a_hr = wmma16(ha1, ldB_t(whhT, 192, c * 16, 32, lane), a_hr);
        a_hz = wmma16(ha0, ldB_t(whhT, 192, 64 + c * 16, 0, lane), a_hz);
        a_hz = wmma16(ha1, ldB_t(whhT, 192, 64 + c * 16, 32, lane), a_hz);
        a_hn = wmma16(ha0, ldB_t(whhT, 192, 128 + c * 16, 0, lane), a_hn);
        a_hn = wmma16(ha1, ldB_t(whhT, 192, 128 + c * 16, 32, lane), a_hn);
        const int col = c * 16 + n;
        const float bir = bih[col],       bhr = bhh[col];
        const float biz = bih[64 + col],  bhz = bhh[64 + col];
        const float bin = bih[128 + col], bhn = bhh[128 + col];
#pragma unroll
        for (int r = 0; r < 8; ++r) {
            const int row = m0 + mh + r;
            const float rg = 1.f / (1.f + __expf(-((a_ir[r] + bir) + (a_hr[r] + bhr))));
            const float zg = 1.f / (1.f + __expf(-((a_iz[r] + biz) + (a_hz[r] + bhz))));
            const float ng = tanhf((a_in[r] + bin) + rg * (a_hn[r] + bhn));
            const float hp = h_in[row * HD + col];
            const float hv = (1.f - zg) * ng + zg * hp;
            out_h[row * HD + col] = hv;
            xs[(mh + r) * HD + col] = (_Float16)hv;
        }
    }

    v16h hf0 = ldA_lds(xs, HD, 0, lane);
    v16h hf1 = ldA_lds(xs, HD, 32, lane);

    // q = h @ fc2_w^T + fc2_b
    {
        v8f acc = {};
        acc = wmma16(hf0, ldB_t(fc2T, NACT_, 0, 0, lane), acc);
        acc = wmma16(hf1, ldB_t(fc2T, NACT_, 0, 32, lane), acc);
        const float bv = fc2_b[n];
#pragma unroll
        for (int r = 0; r < 8; ++r) out_q[(m0 + mh + r) * NACT_ + n] = acc[r] + bv;
    }
    // z1 = h @ emb_w1^T + emb_b1
#pragma unroll
    for (int c = 0; c < 4; ++c) {
        v8f acc = {};
        acc = wmma16(hf0, ldB_t(emb1T, NNH_, c * 16, 0, lane), acc);
        acc = wmma16(hf1, ldB_t(emb1T, NNH_, c * 16, 32, lane), acc);
        const int col = c * 16 + n;
        const float bv = emb_b1[col];
#pragma unroll
        for (int r = 0; r < 8; ++r) z1_buf[(m0 + mh + r) * NNH_ + col] = acc[r] + bv;
    }
}

// ---------------- K2: batchnorm statistics (deterministic column reduction) ----------------

__global__ void __launch_bounds__(256) k2_bnstats(const float* __restrict__ z1, float* __restrict__ stats)
{
    const int col = blockIdx.x;  // 0..63
    float s = 0.f, s2 = 0.f;
    for (int r = threadIdx.x; r < NROWS; r += 256) {
        const float v = z1[r * NNH_ + col];
        s += v; s2 += v * v;
    }
    __shared__ float rs[256], rs2[256];
    rs[threadIdx.x] = s; rs2[threadIdx.x] = s2;
    __syncthreads();
    for (int o = 128; o > 0; o >>= 1) {
        if (threadIdx.x < o) { rs[threadIdx.x] += rs[threadIdx.x + o]; rs2[threadIdx.x] += rs2[threadIdx.x + o]; }
        __syncthreads();
    }
    if (threadIdx.x == 0) {
        const float mean = rs[0] / (float)NROWS;
        const float var = rs2[0] / (float)NROWS - mean * mean;
        stats[col] = mean;
        stats[64 + col] = rsqrtf(var + 1e-5f);
    }
}

// ---------------- K3: bn+lrelu, emb2 GEMM (64->1024), reparameterized latent ----------------

__global__ void __launch_bounds__(256) k3_latent(
    const float* __restrict__ z1, const float* __restrict__ stats,
    const float* __restrict__ bn_g, const float* __restrict__ bn_b,
    const _Float16* __restrict__ emb2T, const float* __restrict__ emb_b2,
    const float* __restrict__ eps, float* __restrict__ lat)
{
    const int lane = threadIdx.x & 31;
    const int wave = threadIdx.x >> 5;
    const int m0 = (blockIdx.x * 8 + wave) * 16;
    const int n = lane & 15;
    const int mh = (lane >> 4) << 3;

    v16h za0, za1;
    {
        const float* zr = z1 + (long)(m0 + (lane & 15)) * NNH_;
        const int kb = (lane >> 4) << 3;
#pragma unroll
        for (int jj = 0; jj < 8; ++jj) {
            const int k0 = kb + jj, k1 = kb + 16 + jj;
            float v0 = (zr[k0]      - stats[k0])      * stats[64 + k0] * bn_g[k0]      + bn_b[k0];
            float v1 = (zr[k0 + 32] - stats[k0 + 32]) * stats[96 + k0] * bn_g[k0 + 32] + bn_b[k0 + 32];
            float v2 = (zr[k1]      - stats[k1])      * stats[64 + k1] * bn_g[k1]      + bn_b[k1];
            float v3 = (zr[k1 + 32] - stats[k1 + 32]) * stats[96 + k1] * bn_g[k1 + 32] + bn_b[k1 + 32];
            za0[jj]     = (_Float16)(v0 > 0.f ? v0 : 0.01f * v0);
            za1[jj]     = (_Float16)(v1 > 0.f ? v1 : 0.01f * v1);
            za0[jj + 8] = (_Float16)(v2 > 0.f ? v2 : 0.01f * v2);
            za1[jj + 8] = (_Float16)(v3 > 0.f ? v3 : 0.01f * v3);
        }
    }
    for (int d = 0; d < 32; ++d) {   // paired tiles: mu at d, logvar at d+32
        v8f am = {}, av = {};
        am = wmma16(za0, ldB_t(emb2T, 1024, d * 16, 0, lane), am);
        am = wmma16(za1, ldB_t(emb2T, 1024, d * 16, 32, lane), am);
        av = wmma16(za0, ldB_t(emb2T, 1024, DLAT + d * 16, 0, lane), av);
        av = wmma16(za1, ldB_t(emb2T, 1024, DLAT + d * 16, 32, lane), av);
        const int co = d * 16 + n;
        const float bm = emb_b2[co], bl = emb_b2[DLAT + co];
#pragma unroll
        for (int r = 0; r < 8; ++r) {
            const long row = m0 + mh + r;
            const float mu = am[r] + bm;
            float va = __expf(av[r] + bl);
            va = va < 0.002f ? 0.002f : va;
            lat[row * DLAT + co] = mu + sqrtf(va) * eps[row * DLAT + co];
        }
    }
}

// ---------------- Kq: WMMA query, folded through wk; qb via cross-lane reduce ----------------

__global__ void __launch_bounds__(256) k_query(
    const float* __restrict__ h, const _Float16* __restrict__ wqT, const float* __restrict__ wq_b,
    const _Float16* __restrict__ wkH, const float* __restrict__ wk_b,
    float* __restrict__ qk, float* __restrict__ qb)
{
    const int lane = threadIdx.x & 31;
    const int wave = threadIdx.x >> 5;
    const int m0 = (blockIdx.x * 8 + wave) * 16;
    const int n = lane & 15;
    const int mh = (lane >> 4) << 3;
    __shared__ _Float16 sh[8][16 * ATT_];
    _Float16* qs = sh[wave];

    v16h ha0 = ldA_f32(h, HD, m0, 0, lane);
    v16h ha1 = ldA_f32(h, HD, m0, 32, lane);
    const float inv = 0.1767766952966369f;  // 1/sqrt(32)

    // query = (h @ wq_w^T + wq_b) * inv : two 16-col tiles
    v8f q0 = {}, q1 = {};
    q0 = wmma16(ha0, ldB_t(wqT, ATT_, 0, 0, lane), q0);
    q0 = wmma16(ha1, ldB_t(wqT, ATT_, 0, 32, lane), q0);
    q1 = wmma16(ha0, ldB_t(wqT, ATT_, 16, 0, lane), q1);
    q1 = wmma16(ha1, ldB_t(wqT, ATT_, 16, 32, lane), q1);
    const float b0 = wq_b[n], b1 = wq_b[16 + n];
    const float wb0 = wk_b[n], wb1 = wk_b[16 + n];
    float qbp[8];
#pragma unroll
    for (int r = 0; r < 8; ++r) {
        const float v0 = (q0[r] + b0) * inv;
        const float v1 = (q1[r] + b1) * inv;
        qs[(mh + r) * ATT_ + n] = (_Float16)v0;
        qs[(mh + r) * ATT_ + 16 + n] = (_Float16)v1;
        qbp[r] = v0 * wb0 + v1 * wb1;   // partial of query . wk_b
    }
    // reduce over the 16 n-lanes of each half-wave (xor masks stay within the half)
#pragma unroll
    for (int off = 1; off < 16; off <<= 1) {
#pragma unroll
        for (int r = 0; r < 8; ++r) qbp[r] += __shfl_xor(qbp[r], off, 32);
    }
    if (n == 0) {
#pragma unroll
        for (int r = 0; r < 8; ++r) qb[m0 + mh + r] = qbp[r];
    }
    // qk = query @ wk_w  (B[k][n] = wk_w[k][n]: wkH is the un-transposed f16 copy)
    v16h qa = ldA_lds(qs, ATT_, 0, lane);
    v8f accqk = {};
    accqk = wmma16(qa, ldB_t(wkH, LD_, 0, 0, lane), accqk);
#pragma unroll
    for (int r = 0; r < 8; ++r) qk[(m0 + mh + r) * LD_ + n] = accqk[r];
}

// ---------------- K4a: msg_h = h @ msg_w1[:, :64]^T + msg_b1 (constant in i) ----------------

__global__ void __launch_bounds__(256) k4a_msgh(
    const float* __restrict__ h, const _Float16* __restrict__ msg1T,
    const float* __restrict__ msg_b1, float* __restrict__ msg_h)
{
    const int lane = threadIdx.x & 31;
    const int wave = threadIdx.x >> 5;
    const int m0 = (blockIdx.x * 8 + wave) * 16;
    const int n = lane & 15;
    const int mh = (lane >> 4) << 3;

    v16h a0 = ldA_f32(h, HD, m0, 0, lane);
    v16h a1 = ldA_f32(h, HD, m0, 32, lane);
#pragma unroll
    for (int c = 0; c < 4; ++c) {
        v8f acc = {};
        acc = wmma16(a0, ldB_t(msg1T, NNH_, c * 16, 0, lane), acc);
        acc = wmma16(a1, ldB_t(msg1T, NNH_, c * 16, 32, lane), acc);
        const int col = c * 16 + n;
        const float bv = msg_b1[col];
#pragma unroll
        for (int r = 0; r < 8; ++r) msg_h[(m0 + mh + r) * NNH_ + col] = acc[r] + bv;
    }
}

// ---------------- K4b: message MLP over 1M (b,i,j) rows ----------------

__global__ void __launch_bounds__(256) k4b_msg(
    const float* __restrict__ lat, const float* __restrict__ msg_h,
    const _Float16* __restrict__ msg1T, const _Float16* __restrict__ msg2T,
    const float* __restrict__ msg_b2, float* __restrict__ msgb)
{
    const int lane = threadIdx.x & 31;
    const int wave = threadIdx.x >> 5;
    const int m0 = (blockIdx.x * 8 + wave) * 16;
    const int n = lane & 15;
    const int mh = (lane >> 4) << 3;
    __shared__ _Float16 sh[8][16 * NNH_];
    _Float16* hs = sh[wave];

    // hid = lrelu(msg_h[b,j] + lat[b,i,j] @ msg_w1[:,64:80]^T)  (K=16 zero-padded to 32)
    v16h la = ldA_lat(lat, m0, lane);
#pragma unroll
    for (int c = 0; c < 4; ++c) {
        v8f acc = {};
        acc = wmma16(la, ldB_t_bk(msg1T, NNH_, c * 16, 64, 80, lane), acc);
        const int col = c * 16 + n;
#pragma unroll
        for (int r = 0; r < 8; ++r) {
            const int row = m0 + mh + r;                     // (b*32+i)*32+j
            const int bj = ((row >> 10) << 5) + (row & 31);  // b*32 + j (h_rep uses agent j)
            float v = acc[r] + msg_h[bj * NNH_ + col];
            hs[(mh + r) * NNH_ + col] = (_Float16)(v > 0.f ? v : 0.01f * v);
        }
    }
    v16h hf0 = ldA_lds(hs, NNH_, 0, lane);
    v16h hf1 = ldA_lds(hs, NNH_, 32, lane);
    v8f acc = {};
    acc = wmma16(hf0, ldB_t(msg2T, NACT_, 0, 0, lane), acc);
    acc = wmma16(hf1, ldB_t(msg2T, NACT_, 0, 32, lane), acc);
    const float bv = msg_b2[n];
#pragma unroll
    for (int r = 0; r < 8; ++r) msgb[(long)(m0 + mh + r) * NACT_ + n] = acc[r] + bv;
}

// ---------------- K5: attention (alpha via qk.lat), softmax, aggregate, output ----------------

__global__ void __launch_bounds__(256) k5_att(
    const float* __restrict__ lat, const float* __restrict__ msgb,
    const float* __restrict__ qk, const float* __restrict__ qb,
    float* __restrict__ out_q)
{
    const int b = blockIdx.x;
    const int t = threadIdx.x;
    __shared__ float alp[AG][AG + 1];

    for (int p = t; p < AG * AG; p += 256) {
        const int i = p >> 5, j = p & 31;
        const int row = b * AG + i;
        const float* lp = lat + ((long)row * AG + j) * LD_;
        const float* qp = qk + row * LD_;
        float s = qb[row];
#pragma unroll
        for (int l = 0; l < LD_; ++l) s += lp[l] * qp[l];
        alp[i][j] = (i == j) ? -1e9f : s;
    }
    __syncthreads();
    if (t < AG) {
        const int i = t;
        float mx = -3.4e38f;
#pragma unroll
        for (int j = 0; j < AG; ++j) mx = fmaxf(mx, alp[i][j]);
        float ssum = 0.f;
#pragma unroll
        for (int j = 0; j < AG; ++j) { const float e = __expf(alp[i][j] - mx); alp[i][j] = e; ssum += e; }
        const float invs = 1.f / ssum;
#pragma unroll
        for (int j = 0; j < AG; ++j) alp[i][j] *= invs;
    }
    __syncthreads();
    for (int p = t; p < AG * NACT_; p += 256) {
        const int j = p >> 4, nn = p & 15;
        float s = 0.f;
#pragma unroll
        for (int i = 0; i < AG; ++i)
            s += alp[i][j] * msgb[(((long)(b * AG + i)) * AG + j) * NACT_ + nn];
        const int orow = b * AG + j;
        out_q[orow * NACT_ + nn] += s;   // q written by K1, agg added here
    }
}

// ---------------- launch ----------------

extern "C" void kernel_launch(void* const* d_in, const int* in_sizes, int n_in,
                              void* d_out, int out_size, void* d_ws, size_t ws_size,
                              hipStream_t stream)
{
    (void)in_sizes; (void)n_in; (void)out_size; (void)ws_size;
    const float* inputs = (const float*)d_in[0];
    const float* hidden = (const float*)d_in[1];
    const float* eps    = (const float*)d_in[2];
    const float* fc1_w  = (const float*)d_in[3];
    const float* fc1_b  = (const float*)d_in[4];
    const float* wih    = (const float*)d_in[5];
    const float* whh    = (const float*)d_in[6];
    const float* bih    = (const float*)d_in[7];
    const float* bhh    = (const float*)d_in[8];
    const float* fc2_w  = (const float*)d_in[9];
    const float* fc2_b  = (const float*)d_in[10];
    const float* emb_w1 = (const float*)d_in[11];
    const float* emb_b1 = (const float*)d_in[12];
    const float* bn_g   = (const float*)d_in[13];
    const float* bn_b   = (const float*)d_in[14];
    const float* emb_w2 = (const float*)d_in[15];
    const float* emb_b2 = (const float*)d_in[16];
    const float* msg_w1 = (const float*)d_in[17];
    const float* msg_b1 = (const float*)d_in[18];
    const float* msg_w2 = (const float*)d_in[19];
    const float* msg_b2 = (const float*)d_in[20];
    const float* wq_w   = (const float*)d_in[21];
    const float* wq_b   = (const float*)d_in[22];
    const float* wk_w   = (const float*)d_in[23];
    const float* wk_b   = (const float*)d_in[24];

    float* out_q = (float*)d_out;                 // 32768 x 16
    float* out_h = out_q + NROWS * NACT_;         // 32768 x 64

    float* ws     = (float*)d_ws;
    float* z1_buf = ws;                           // 32768*64
    float* stats  = z1_buf + NROWS * NNH_;        // 128 (mean, inv_std)
    float* latb   = stats + 128;                  // 32768*512
    float* msg_h  = latb + (long)NROWS * DLAT;    // 32768*64
    float* msgb   = msg_h + NROWS * NNH_;         // 1048576*16
    float* qkb    = msgb + (long)M2ROWS * NACT_;  // 32768*16
    float* qbb    = qkb + NROWS * LD_;            // 32768

    // f16 transposed weight region (after the float region)
    _Float16* wt   = (_Float16*)(qbb + NROWS);
    _Float16* fc1T  = wt;                 // 96*64   = 6144
    _Float16* wihT  = fc1T + 6144;        // 64*192  = 12288
    _Float16* whhT  = wihT + 12288;       // 64*192  = 12288
    _Float16* fc2T  = whhT + 12288;       // 64*16   = 1024
    _Float16* emb1T = fc2T + 1024;        // 64*64   = 4096
    _Float16* emb2T = emb1T + 4096;       // 64*1024 = 65536
    _Float16* msg1T = emb2T + 65536;      // 80*64   = 5120
    _Float16* msg2T = msg1T + 5120;       // 64*16   = 1024
    _Float16* wqT   = msg2T + 1024;       // 64*32   = 2048
    _Float16* wkH   = wqT + 2048;         // 32*16   = 512 (un-transposed)

    // repack all weights to f16 (once per launch; L2-resident afterwards)
    k0_repack<<<24, 256, 0, stream>>>(fc1_w, fc1T, HD, IND);
    k0_repack<<<48, 256, 0, stream>>>(wih, wihT, 192, HD);
    k0_repack<<<48, 256, 0, stream>>>(whh, whhT, 192, HD);
    k0_repack<<<4, 256, 0, stream>>>(fc2_w, fc2T, NACT_, HD);
    k0_repack<<<16, 256, 0, stream>>>(emb_w1, emb1T, NNH_, HD);
    k0_repack<<<256, 256, 0, stream>>>(emb_w2, emb2T, 1024, NNH_);
    k0_repack<<<20, 256, 0, stream>>>(msg_w1, msg1T, NNH_, 80);
    k0_repack<<<4, 256, 0, stream>>>(msg_w2, msg2T, NACT_, HD);
    k0_repack<<<8, 256, 0, stream>>>(wq_w, wqT, ATT_, HD);
    k0_tof16<<<2, 256, 0, stream>>>(wk_w, wkH, ATT_ * LD_);

    k1_fc1_gru<<<NROWS / 128, 256, 0, stream>>>(inputs, hidden, fc1T, fc1_b, wihT, whhT, bih, bhh,
                                                fc2T, fc2_b, emb1T, emb_b1, out_q, out_h, z1_buf);
    k2_bnstats<<<NNH_, 256, 0, stream>>>(z1_buf, stats);
    k3_latent<<<NROWS / 128, 256, 0, stream>>>(z1_buf, stats, bn_g, bn_b, emb2T, emb_b2, eps, latb);
    k_query<<<NROWS / 128, 256, 0, stream>>>(out_h, wqT, wq_b, wkH, wk_b, qkb, qbb);
    k4a_msgh<<<NROWS / 128, 256, 0, stream>>>(out_h, msg1T, msg_b1, msg_h);
    k4b_msg<<<M2ROWS / 128, 256, 0, stream>>>(latb, msg_h, msg1T, msg2T, msg_b2, msgb);
    k5_att<<<BS_, 256, 0, stream>>>(latb, msgb, qkb, qbb, out_q);
}
